// SelfCompressingRGCNAutoEncoder_27917287424629
// MI455X (gfx1250) — matrix operand order repaired and checked by reading, not compiled, not run on hardware
//
#include <hip/hip_runtime.h>

typedef unsigned short u16;
typedef unsigned int   u32;

typedef __attribute__((ext_vector_type(16))) __bf16 v16bf;
typedef __attribute__((ext_vector_type(8)))  __bf16 v8bf;
typedef __attribute__((ext_vector_type(8)))  float  v8f;

#define FD 128   // F (input feature dim)
#define HD 128   // H (hidden dim)
#define CD 64    // C (clusters)

// ---------- helpers ----------
static __device__ __forceinline__ float bf2f(u16 u) {
  union { float f; u32 i; } v; v.i = ((u32)u) << 16; return v.f;
}
static __device__ __forceinline__ u16 f2bf(float f) {
  union { float f; u32 i; } v; v.f = f;
  u32 i = v.i + 0x7FFFu + ((v.i >> 16) & 1u);   // round-to-nearest-even
  return (u16)(i >> 16);
}
static __device__ __forceinline__ float sigm(float x) { return 1.0f / (1.0f + __expf(-x)); }
static __device__ __forceinline__ float hardconc(float la) {
  // TEMP=2/3 -> *1.5 ; STRETCH=0.1 ; CLIP=2
  float pre = fminf(fmaxf(la * 1.5f, -2.0f), 2.0f);
  float z = sigm(pre) * 1.2f - 0.1f;
  return fminf(fmaxf(z, 0.0f), 1.0f);
}

// ---------- K0: convert x and weights to bf16 (weights pre-transposed) ----------
__global__ void k_convert(const float* __restrict__ x, const float* __restrict__ wself,
                          const float* __restrict__ wrel, const float* __restrict__ aw,
                          u16* __restrict__ xb, u16* __restrict__ wst,
                          u16* __restrict__ wrt, u16* __restrict__ awt,
                          int N, int R)
{
  int stride = gridDim.x * blockDim.x;
  int t0 = blockIdx.x * blockDim.x + threadIdx.x;
  for (int i = t0; i < N * FD; i += stride) xb[i] = f2bf(x[i]);
  for (int i = t0; i < FD * HD; i += stride) {            // wst[h][f] = W_self[f][h]
    int f = i >> 7, h = i & 127;
    wst[h * FD + f] = f2bf(wself[i]);
  }
  for (int i = t0; i < R * FD * HD; i += stride) {        // wrt[r][h][f] = W_rel[r][f][h]
    int r = i >> 14, rem = i & 16383, f = rem >> 7, h = rem & 127;
    wrt[(r << 14) + h * FD + f] = f2bf(wrel[i]);
  }
  for (int i = t0; i < HD * CD; i += stride) {            // awt[c][h] = assign_W[h][c]
    int h = i >> 6, c = i & 63;
    awt[c * HD + h] = f2bf(aw[i]);
  }
}

// ---------- K2: per (dst, relation) degree histogram ----------
__global__ void k_deg(const int* __restrict__ dst, const int* __restrict__ et,
                      u32* __restrict__ deg, int E, int R)
{
  int i = blockIdx.x * blockDim.x + threadIdx.x;
  if (i < E) atomicAdd(&deg[(size_t)dst[i] * R + et[i]], 1u);
}

// ---------- K3: h_all[r] = x @ W_rel[r]  (bf16 WMMA, one wave per 16x128 strip) ----------
__global__ void __launch_bounds__(256) k_relgemm(
    const u16* __restrict__ xb, const u16* __restrict__ wrt,
    u16* __restrict__ hall, int N, int R, int spr)
{
  int wave = blockIdx.x * (blockDim.x >> 5) + (threadIdx.x >> 5);
  int lane = threadIdx.x & 31;
  if (wave >= R * spr) return;
  int r = wave / spr;
  int row0 = (wave - r * spr) * 16;
  int half = lane >> 4;
  int nl = lane & 15;
  int arow = row0 + nl;
  if (arow >= N) arow = N - 1;
  const u16* xrow = xb + (size_t)arow * FD;
  const u16* wb = wrt + ((size_t)r << 14);

  v8f acc[8];
#pragma unroll
  for (int t = 0; t < 8; t++) acc[t] = (v8f){0.f,0.f,0.f,0.f,0.f,0.f,0.f,0.f};

#pragma unroll
  for (int k0 = 0; k0 < FD; k0 += 32) {
    v8bf alo = *(const v8bf*)(xrow + k0 + 16 * half);
    v8bf ahi = *(const v8bf*)(xrow + k0 + 16 * half + 8);
    v16bf a;
#pragma unroll
    for (int i = 0; i < 8; i++) { a[i] = alo[i]; a[8 + i] = ahi[i]; }
    v16bf bt[8];
#pragma unroll
    for (int t = 0; t < 8; t++)
      bt[t] = *(const v16bf*)(wb + (size_t)(t * 16 + nl) * FD + k0 + 16 * half);
#pragma unroll
    for (int t = 0; t < 8; t++)
      acc[t] = __builtin_amdgcn_wmma_f32_16x16x32_bf16(false, a, false, bt[t],
                                                       (short)0, acc[t], false, false);
  }

  u16* out = hall + ((size_t)r * N + row0) * HD;
#pragma unroll
  for (int t = 0; t < 8; t++) {
    int col = t * 16 + nl;
#pragma unroll
    for (int j = 0; j < 8; j++) {
      int m = j + 8 * half;
      if (row0 + m < N) out[(size_t)m * HD + col] = f2bf(acc[t][j]);
    }
  }
}

// ---------- K4: gather message, scale by 1/deg, atomic scatter into agg ----------
__global__ void __launch_bounds__(256) k_msg(
    const int* __restrict__ src, const int* __restrict__ dst, const int* __restrict__ et,
    const u16* __restrict__ hall, const u32* __restrict__ deg,
    float* __restrict__ agg, int N, int R, int E)
{
  int e = blockIdx.x * (blockDim.x >> 5) + (threadIdx.x >> 5);
  int lane = threadIdx.x & 31;
  if (e >= E) return;
  int s = src[e], d = dst[e], r = et[e];
  u32 c = deg[(size_t)d * R + r];
  float norm = 1.0f / fmaxf((float)c, 1.0f);
  const u16* hp = hall + ((size_t)r * N + s) * HD + lane * 4;
  uint2 u = *(const uint2*)hp;
  float v0 = bf2f((u16)(u.x & 0xffffu)) * norm;
  float v1 = bf2f((u16)(u.x >> 16)) * norm;
  float v2 = bf2f((u16)(u.y & 0xffffu)) * norm;
  float v3 = bf2f((u16)(u.y >> 16)) * norm;
  float* ap = agg + (size_t)d * HD + lane * 4;
  unsafeAtomicAdd(ap + 0, v0);
  unsafeAtomicAdd(ap + 1, v1);
  unsafeAtomicAdd(ap + 2, v2);
  unsafeAtomicAdd(ap + 3, v3);
}

// ---------- K5: h = relu(agg + x @ W_self) ----------
__global__ void __launch_bounds__(256) k_selfgemm(
    const u16* __restrict__ xb, const u16* __restrict__ wst,
    const float* __restrict__ agg, u16* __restrict__ hb, int N, int spr)
{
  int wave = blockIdx.x * (blockDim.x >> 5) + (threadIdx.x >> 5);
  int lane = threadIdx.x & 31;
  if (wave >= spr) return;
  int row0 = wave * 16;
  int half = lane >> 4;
  int nl = lane & 15;
  int arow = row0 + nl;
  if (arow >= N) arow = N - 1;
  const u16* xrow = xb + (size_t)arow * FD;

  v8f acc[8];
#pragma unroll
  for (int t = 0; t < 8; t++) acc[t] = (v8f){0.f,0.f,0.f,0.f,0.f,0.f,0.f,0.f};

#pragma unroll
  for (int k0 = 0; k0 < FD; k0 += 32) {
    v8bf alo = *(const v8bf*)(xrow + k0 + 16 * half);
    v8bf ahi = *(const v8bf*)(xrow + k0 + 16 * half + 8);
    v16bf a;
#pragma unroll
    for (int i = 0; i < 8; i++) { a[i] = alo[i]; a[8 + i] = ahi[i]; }
    v16bf bt[8];
#pragma unroll
    for (int t = 0; t < 8; t++)
      bt[t] = *(const v16bf*)(wst + (size_t)(t * 16 + nl) * FD + k0 + 16 * half);
#pragma unroll
    for (int t = 0; t < 8; t++)
      acc[t] = __builtin_amdgcn_wmma_f32_16x16x32_bf16(false, a, false, bt[t],
                                                       (short)0, acc[t], false, false);
  }

#pragma unroll
  for (int t = 0; t < 8; t++) {
    int col = t * 16 + nl;
#pragma unroll
    for (int j = 0; j < 8; j++) {
      int row = row0 + j + 8 * half;
      if (row < N) {
        float v = acc[t][j] + agg[(size_t)row * HD + col];
        hb[(size_t)row * HD + col] = f2bf(fmaxf(v, 0.0f));
      }
    }
  }
}

// ---------- K7: gated inter-cluster weights (interleaved bf16 layout) + cluster gate ----
// layout: wT[r*4096 + (cp>>1)*128 + c*2 + (cp&1)]  -> one b64/lane serves 2 cp steps
__global__ void k_weights(const float* __restrict__ il, const float* __restrict__ igla,
                          const float* __restrict__ cla, u16* __restrict__ wT,
                          float* __restrict__ cgate, int R)
{
  int i = blockIdx.x * blockDim.x + threadIdx.x;
  int total = R * CD * CD;
  if (i < total) {
    int r = i / (CD * CD);
    int rem = i - r * CD * CD;
    int c = rem >> 6, cp = rem & 63;
    float w = sigm(il[i]) * hardconc(igla[i]);
    wT[(size_t)r * (CD * CD) + (cp >> 1) * 128 + c * 2 + (cp & 1)] = f2bf(w);
  }
  if (i < CD) cgate[i] = hardconc(cla[i]);
}

// ---------- K6: assign = softmax(h @ assign_W + b) * cluster_gate ----------
__global__ void __launch_bounds__(256) k_assign(
    const u16* __restrict__ hb, const u16* __restrict__ awt,
    const float* __restrict__ ab, const float* __restrict__ cgate,
    u16* __restrict__ asg, int N, int spr)
{
  int wave = blockIdx.x * (blockDim.x >> 5) + (threadIdx.x >> 5);
  int lane = threadIdx.x & 31;
  if (wave >= spr) return;
  int row0 = wave * 16;
  int half = lane >> 4;
  int nl = lane & 15;
  int arow = row0 + nl;
  if (arow >= N) arow = N - 1;
  const u16* hrow = hb + (size_t)arow * HD;

  v8f acc[4];
#pragma unroll
  for (int t = 0; t < 4; t++) acc[t] = (v8f){0.f,0.f,0.f,0.f,0.f,0.f,0.f,0.f};

#pragma unroll
  for (int k0 = 0; k0 < HD; k0 += 32) {
    v8bf alo = *(const v8bf*)(hrow + k0 + 16 * half);
    v8bf ahi = *(const v8bf*)(hrow + k0 + 16 * half + 8);
    v16bf a;
#pragma unroll
    for (int i = 0; i < 8; i++) { a[i] = alo[i]; a[8 + i] = ahi[i]; }
    v16bf bt[4];
#pragma unroll
    for (int t = 0; t < 4; t++)
      bt[t] = *(const v16bf*)(awt + (size_t)(t * 16 + nl) * HD + k0 + 16 * half);
#pragma unroll
    for (int t = 0; t < 4; t++)
      acc[t] = __builtin_amdgcn_wmma_f32_16x16x32_bf16(false, a, false, bt[t],
                                                       (short)0, acc[t], false, false);
  }

  float bb[4], gg[4];
#pragma unroll
  for (int t = 0; t < 4; t++) { bb[t] = ab[t * 16 + nl]; gg[t] = cgate[t * 16 + nl]; }

#pragma unroll
  for (int j = 0; j < 8; j++) {
    float v0 = acc[0][j] + bb[0];
    float v1 = acc[1][j] + bb[1];
    float v2 = acc[2][j] + bb[2];
    float v3 = acc[3][j] + bb[3];
    float mx = fmaxf(fmaxf(v0, v1), fmaxf(v2, v3));
#pragma unroll
    for (int m = 1; m <= 8; m <<= 1) mx = fmaxf(mx, __shfl_xor(mx, m, 32));
    float e0 = __expf(v0 - mx), e1 = __expf(v1 - mx);
    float e2 = __expf(v2 - mx), e3 = __expf(v3 - mx);
    float sm = e0 + e1 + e2 + e3;
#pragma unroll
    for (int m = 1; m <= 8; m <<= 1) sm += __shfl_xor(sm, m, 32);
    float inv = 1.0f / sm;
    int row = row0 + j + 8 * half;
    if (row < N) {
      u16* arw = asg + (size_t)row * CD + nl;
      arw[0]  = f2bf(e0 * inv * gg[0]);
      arw[16] = f2bf(e1 * inv * gg[1]);
      arw[32] = f2bf(e2 * inv * gg[2]);
      arw[48] = f2bf(e3 * inv * gg[3]);
    }
  }
}

// ---------- K8: per-edge bilinear  logit = src_a . (W[r] dst_a) + bias[r] ----------
__global__ void __launch_bounds__(256) k_edge(
    const int* __restrict__ src, const int* __restrict__ dst, const int* __restrict__ et,
    const u16* __restrict__ asg, const u16* __restrict__ wT,
    const float* __restrict__ abias, float* __restrict__ out, int E, int R)
{
  __shared__ u16 lw[8 * CD * CD];          // 64 KB: all relations' gated weights (bf16)
  int tid = threadIdx.x;
  int nqw = (R * CD * CD) >> 2;            // uint2 (4 bf16) chunks
  for (int i = tid; i < nqw; i += blockDim.x) ((uint2*)lw)[i] = ((const uint2*)wT)[i];
  __syncthreads();

  int lane = tid & 31;
  int wpb = blockDim.x >> 5;
  int totalWaves = gridDim.x * wpb;
  for (int e = blockIdx.x * wpb + (tid >> 5); e < E; e += totalWaves) {
    int s = src[e], d = dst[e], r = et[e];
    u32 su = *(const u32*)(asg + (size_t)s * CD + 2 * lane);
    u32 du = *(const u32*)(asg + (size_t)d * CD + 2 * lane);
    float s0 = bf2f((u16)(su & 0xffffu)), s1 = bf2f((u16)(su >> 16));
    float d0 = bf2f((u16)(du & 0xffffu)), d1 = bf2f((u16)(du >> 16));
    float acc0 = 0.f, acc1 = 0.f;
    const u16* wr = lw + r * (CD * CD);
#pragma unroll 8
    for (int cp2 = 0; cp2 < (CD >> 1); ++cp2) {
      // dst_a[2*cp2] and dst_a[2*cp2+1] live in lane cp2 (.x / .y)
      float dva = __shfl(d0, cp2, 32);
      float dvb = __shfl(d1, cp2, 32);
      // one b64 per lane: (c=2l, cp even), (c=2l, cp odd), (c=2l+1, cp even), (c=2l+1, cp odd)
      uint2 w = *(const uint2*)(wr + cp2 * 128 + 4 * lane);
      acc0 += bf2f((u16)(w.x & 0xffffu)) * dva;
      acc0 += bf2f((u16)(w.x >> 16)) * dvb;
      acc1 += bf2f((u16)(w.y & 0xffffu)) * dva;
      acc1 += bf2f((u16)(w.y >> 16)) * dvb;
    }
    float p = s0 * acc0 + s1 * acc1;
#pragma unroll
    for (int m = 16; m >= 1; m >>= 1) p += __shfl_xor(p, m, 32);
    if (lane == 0) out[e] = p + abias[r];
  }
}

// ---------- host-side orchestration ----------
extern "C" void kernel_launch(void* const* d_in, const int* in_sizes, int n_in,
                              void* d_out, int out_size, void* d_ws, size_t ws_size,
                              hipStream_t stream)
{
  const float* x     = (const float*)d_in[0];
  const int*   ei    = (const int*)d_in[1];
  const int*   et    = (const int*)d_in[2];
  const float* wself = (const float*)d_in[3];
  const float* wrel  = (const float*)d_in[4];
  const float* aw    = (const float*)d_in[5];
  const float* ab    = (const float*)d_in[6];
  const float* cla   = (const float*)d_in[7];
  const float* il    = (const float*)d_in[8];
  const float* igla  = (const float*)d_in[9];
  const float* abias = (const float*)d_in[10];

  int E = in_sizes[2];
  int R = in_sizes[10];
  int N = in_sizes[0] / FD;
  const int* srcp = ei;
  const int* dstp = ei + E;

  char* ws = (char*)d_ws;
  size_t off = 0;
  auto alloc = [&](size_t bytes) -> size_t {
    size_t p = off; off += (bytes + 255) & ~(size_t)255; return p;
  };
  size_t o_xb   = alloc((size_t)N * FD * 2);
  size_t o_wrt  = alloc((size_t)R * FD * HD * 2);
  size_t o_wst  = alloc((size_t)FD * HD * 2);
  size_t o_awt  = alloc((size_t)HD * CD * 2);
  size_t o_hall = alloc((size_t)R * N * HD * 2);
  size_t o_deg  = alloc((size_t)N * R * 4);
  size_t o_agg  = alloc((size_t)N * HD * 4);
  size_t o_hb   = alloc((size_t)N * HD * 2);
  size_t o_asg  = alloc((size_t)N * CD * 2);
  size_t o_wT   = alloc((size_t)R * CD * CD * 2);
  size_t o_cg   = alloc((size_t)CD * 4);
  (void)ws_size; (void)n_in; (void)out_size;

  u16*  xb    = (u16*)(ws + o_xb);
  u16*  wrt   = (u16*)(ws + o_wrt);
  u16*  wst   = (u16*)(ws + o_wst);
  u16*  awt   = (u16*)(ws + o_awt);
  u16*  hall  = (u16*)(ws + o_hall);
  u32*  deg   = (u32*)(ws + o_deg);
  float* agg  = (float*)(ws + o_agg);
  u16*  hb    = (u16*)(ws + o_hb);
  u16*  asg   = (u16*)(ws + o_asg);
  u16*  wT    = (u16*)(ws + o_wT);
  float* cg   = (float*)(ws + o_cg);

  hipMemsetAsync(ws + o_deg, 0, (size_t)N * R * 4, stream);
  hipMemsetAsync(ws + o_agg, 0, (size_t)N * HD * 4, stream);

  int spr = (N + 15) / 16;

  k_convert<<<4096, 256, 0, stream>>>(x, wself, wrel, aw, xb, wst, wrt, awt, N, R);
  k_deg<<<(E + 255) / 256, 256, 0, stream>>>(dstp, et, deg, E, R);
  int w3 = R * spr;
  k_relgemm<<<(w3 + 7) / 8, 256, 0, stream>>>(xb, wrt, hall, N, R, spr);
  k_msg<<<(E + 7) / 8, 256, 0, stream>>>(srcp, dstp, et, hall, deg, agg, N, R, E);
  k_selfgemm<<<(spr + 7) / 8, 256, 0, stream>>>(xb, wst, agg, hb, N, spr);
  k_weights<<<(R * CD * CD + 255) / 256, 256, 0, stream>>>(il, igla, cla, wT, cg, R);
  k_assign<<<(spr + 7) / 8, 256, 0, stream>>>(hb, awt, ab, cg, asg, N, spr);
  k_edge<<<4096, 256, 0, stream>>>(srcp, dstp, et, asg, wT, abias, (float*)d_out, E, R);
}
